// Mamba2_21715354648630
// MI455X (gfx1250) — compile-verified, matrix operand order checked by reading, not run on hardware
//
#include <hip/hip_runtime.h>
#include <hip/hip_bf16.h>

// ---------------- problem constants ----------------
constexpr int NB     = 2;
constexpr int LSEQ   = 4096;
constexpr int DM     = 1024;
constexpr int DI     = 2048;
constexpr int DSTATE = 128;
constexpr int NH     = 16;
constexpr int HD     = 128;
constexpr int CH     = 256;
constexpr int NCH    = LSEQ / CH;           // 16
constexpr int CONVD  = DI + 2 * DSTATE;     // 2304
constexpr int NPROJ  = 2 * DI + 2 * DSTATE + NH; // 4368

// ---------------- WMMA types ----------------
typedef __attribute__((ext_vector_type(16))) __bf16 v16bf;
typedef __attribute__((ext_vector_type(8)))  __bf16 v8bf;
typedef __attribute__((ext_vector_type(8)))  float  v8f;
// 16-byte chunk type matching the async-to-LDS builtin's parameter type
typedef int v4i_b __attribute__((vector_size(16)));

// LDS tile row stride (bf16 elements). 40*2 = 80 bytes: 16B aligned rows.
constexpr int TSTR = 40;

// ---------------- async copy to LDS (CDNA5) ----------------
#if defined(__has_builtin)
#if __has_builtin(__builtin_amdgcn_global_load_async_to_lds_b128)
#define HAVE_ASYNC_COPY 1
#endif
#if __has_builtin(__builtin_amdgcn_s_wait_asynccnt)
#define HAVE_WAIT_ASYNC 1
#endif
#endif
#ifndef HAVE_ASYNC_COPY
#define HAVE_ASYNC_COPY 0
#endif
#ifndef HAVE_WAIT_ASYNC
#define HAVE_WAIT_ASYNC 0
#endif

// copy 16 bytes global -> LDS
__device__ inline void stage16(const __bf16* g, __bf16* l) {
#if HAVE_ASYNC_COPY
  __builtin_amdgcn_global_load_async_to_lds_b128((v4i_b*)g, (v4i_b*)l, 0, 0);
#else
  *(uint4*)l = *(const uint4*)g;
#endif
}

__device__ inline void wait_async0() {
#if HAVE_ASYNC_COPY
#if HAVE_WAIT_ASYNC
  __builtin_amdgcn_s_wait_asynccnt(0);
#else
  asm volatile("s_wait_asynccnt 0x0" ::: "memory");
#endif
#endif
}

// Load one 16x32 bf16 fragment row for this lane from an LDS tile.
// Per ISA 7.12.2: lane l holds row (l&15); element i <-> k = i + 8*(l>>4) + 8*(i>=8).
__device__ inline v16bf ld_frag(const __bf16* rowBase, int hh) {
  v8bf lo = *(const v8bf*)(rowBase + 8 * hh);
  v8bf hi = *(const v8bf*)(rowBase + 16 + 8 * hh);
  v16bf o;
#pragma unroll
  for (int i = 0; i < 8; ++i) { o[i] = lo[i]; o[i + 8] = hi[i]; }
  return o;
}

__device__ inline v8f wmma_bf16(v16bf a, v16bf b, v8f c) {
  return __builtin_amdgcn_wmma_f32_16x16x32_bf16(
      false, a, false, b, (short)0, c, false, false);
}

__device__ inline float silu_f(float x) { return x / (1.f + __expf(-x)); }

// =====================================================================
// fp32 -> bf16 bulk conversion (4 elements/thread)
// =====================================================================
__global__ void f32_to_bf16_kernel(const float* __restrict__ src,
                                   __bf16* __restrict__ dst, long n) {
  const long i = ((long)blockIdx.x * blockDim.x + threadIdx.x) * 4;
  if (i + 3 >= n) {
    for (long j = i; j < n; ++j) dst[j] = (__bf16)src[j];
    return;
  }
  const float4 v = *(const float4*)(src + i);
  dst[i + 0] = (__bf16)v.x; dst[i + 1] = (__bf16)v.y;
  dst[i + 2] = (__bf16)v.z; dst[i + 3] = (__bf16)v.w;
}

// =====================================================================
// Generic batched GEMM:  C[M,N] = A[M,K] * B[N,K]^T   (bf16 in, fp32 out)
// 256 threads (8 waves), 128x128 tile, K stepped by 32, double-buffered
// LDS with async global->LDS copies overlapping WMMA compute.
// Loads are unguarded: callers pad operand buffers past M/N as needed.
// =====================================================================
__global__ void gemm_nt_bf16(const __bf16* __restrict__ A, int lda, long sAz,
                             const __bf16* __restrict__ Bw, int ldb, long sBz,
                             float* __restrict__ C, int ldc, long sCz,
                             int M, int N, int K) {
  __shared__ __align__(16) __bf16 As[2][128 * TSTR];
  __shared__ __align__(16) __bf16 Bs[2][128 * TSTR];
  const int z = blockIdx.z;
  A += (long)z * sAz; Bw += (long)z * sBz; C += (long)z * sCz;
  const int n0 = blockIdx.x * 128, m0 = blockIdx.y * 128;
  const int t = threadIdx.x;
  const int wave = t >> 5, lane = t & 31, r = lane & 15, hh = lane >> 4;
  const int wm = (wave & 3) * 32;   // wave's row offset in block tile
  const int wn = (wave >> 2) * 64;  // wave's col offset in block tile

  // staging map: 2 threads per row, 2x16B chunks each (row = 32 bf16 = 64B)
  const int srow  = t >> 1;
  const int scole = (t & 1) * 16;   // bf16 element offset within the slab
  const __bf16* gA = A + (long)(m0 + srow) * lda + scole;
  const __bf16* gB = Bw + (long)(n0 + srow) * ldb + scole;
  const int ldsOff = srow * TSTR + scole;

  v8f acc[2][4];
#pragma unroll
  for (int i = 0; i < 2; ++i)
#pragma unroll
    for (int j = 0; j < 4; ++j)
#pragma unroll
      for (int e = 0; e < 8; ++e) acc[i][j][e] = 0.f;

  const int NK = K / 32;
  // prologue: stage slab 0 into buffer 0
  stage16(gA, &As[0][ldsOff]);
  stage16(gA + 8, &As[0][ldsOff + 8]);
  stage16(gB, &Bs[0][ldsOff]);
  stage16(gB + 8, &Bs[0][ldsOff + 8]);

  int buf = 0;
  for (int ki = 0; ki < NK; ++ki) {
    wait_async0();        // this wave's current-slab copies have landed
    __syncthreads();      // everyone's copies landed; prior reads retired
    if (ki + 1 < NK) {    // stage next slab into the other buffer (overlaps)
      const int k0n = (ki + 1) * 32;
      stage16(gA + k0n, &As[buf ^ 1][ldsOff]);
      stage16(gA + k0n + 8, &As[buf ^ 1][ldsOff + 8]);
      stage16(gB + k0n, &Bs[buf ^ 1][ldsOff]);
      stage16(gB + k0n + 8, &Bs[buf ^ 1][ldsOff + 8]);
    }
    // compute current slab
    v16bf af[2];
#pragma unroll
    for (int i = 0; i < 2; ++i)
      af[i] = ld_frag(&As[buf][(wm + i * 16 + r) * TSTR], hh);
#pragma unroll
    for (int j = 0; j < 4; ++j) {
      v16bf bf_ = ld_frag(&Bs[buf][(wn + j * 16 + r) * TSTR], hh);
#pragma unroll
      for (int i = 0; i < 2; ++i)
        acc[i][j] = wmma_bf16(af[i], bf_, acc[i][j]);
    }
    buf ^= 1;
  }
#pragma unroll
  for (int i = 0; i < 2; ++i) {
#pragma unroll
    for (int j = 0; j < 4; ++j) {
      const int n = n0 + wn + j * 16 + r;
      if (n < N) {
#pragma unroll
        for (int e = 0; e < 8; ++e) {
          const int m = m0 + wm + i * 16 + e + 8 * hh;
          if (m < M) C[(long)m * ldc + n] = acc[i][j][e];
        }
      }
    }
  }
}

// =====================================================================
// Depthwise causal conv(4) + bias + SiLU; writes fp32 and bf16 copies.
// =====================================================================
__global__ void conv_silu_kernel(const float* __restrict__ zx,
                                 const float* __restrict__ cw,
                                 const float* __restrict__ cb,
                                 float* __restrict__ xbc,
                                 __bf16* __restrict__ xbc_bf) {
  long idx = (long)blockIdx.x * blockDim.x + threadIdx.x;
  const long total = (long)NB * LSEQ * CONVD;
  if (idx >= total) return;
  const int c = (int)(idx % CONVD);
  const long bl = idx / CONVD;
  const int l = (int)(bl % LSEQ);
  const int b = (int)(bl / LSEQ);
  const float* base = zx + ((long)b * LSEQ) * NPROJ + DI + c;
  float acc = cb[c];
#pragma unroll
  for (int j = 0; j < 4; ++j) {
    const int ll = l - 3 + j;
    if (ll >= 0) acc += cw[c * 4 + j] * base[(long)ll * NPROJ];
  }
  const float s = silu_f(acc);
  const long o = ((long)b * LSEQ + l) * CONVD + c;
  xbc[o] = s;
  xbc_bf[o] = (__bf16)s;
}

// =====================================================================
// dt = softplus(raw + bias); per-chunk cumsum of dt*A; chunk decay.
// =====================================================================
__global__ void dt_cumsum_kernel(const float* __restrict__ zx,
                                 const float* __restrict__ dt_bias,
                                 const float* __restrict__ A_log,
                                 float* __restrict__ dtbuf,
                                 float* __restrict__ acs,
                                 float* __restrict__ cdec) {
  const int tid = blockIdx.x * blockDim.x + threadIdx.x;
  if (tid >= NB * NCH * NH) return;
  const int h = tid % NH;
  const int c = (tid / NH) % NCH;
  const int b = tid / (NH * NCH);
  const float Ah = -__expf(A_log[h]);
  float cum = 0.f;
  const long obase = ((long)(b * NCH + c) * NH + h) * CH;
  for (int q = 0; q < CH; ++q) {
    const long l = (long)c * CH + q;
    const float raw = zx[((long)b * LSEQ + l) * NPROJ + DI + CONVD + h] + dt_bias[h];
    const float dtv = (raw > 20.f) ? raw : log1pf(__expf(raw));
    dtbuf[obase + q] = dtv;
    cum += dtv * Ah;
    acs[obase + q] = cum;
  }
  cdec[(b * NCH + c) * NH + h] = __expf(cum);
}

// =====================================================================
// Y_diag[q,p] = sum_s (CB[q,s] * Lmat[q,s]) * (x[s,p]*dt[s]),  K = s = 256
// =====================================================================
__global__ void ydiag_kernel(const float* __restrict__ cbuf,
                             const float* __restrict__ xbc,
                             const float* __restrict__ dtbuf,
                             const float* __restrict__ acs,
                             float* __restrict__ ybuf) {
  __shared__ __align__(16) __bf16 As[64 * TSTR];
  __shared__ __align__(16) __bf16 Bs[64 * TSTR];
  const int z = blockIdx.z;
  const int h = z % NH, bc = z / NH;
  const int q0 = blockIdx.y * 64, p0 = blockIdx.x * 64;
  const float* CB   = cbuf + (long)bc * CH * CH;
  const float* acsh = acs   + ((long)bc * NH + h) * CH;
  const float* dth  = dtbuf + ((long)bc * NH + h) * CH;
  const float* xh   = xbc + (long)bc * CH * CONVD + h * HD;
  const int t = threadIdx.x;
  const int wave = t >> 5, lane = t & 31, r = lane & 15, hh = lane >> 4;

  v8f acc[4];
#pragma unroll
  for (int j = 0; j < 4; ++j)
#pragma unroll
    for (int e = 0; e < 8; ++e) acc[j][e] = 0.f;

  for (int s0 = 0; s0 < CH; s0 += 32) {
#pragma unroll
    for (int ph = 0; ph < 2; ++ph) {
      const int row = (t >> 2) + ph * 32;
      const int col = (t & 3) * 8;
      const int q = q0 + row;
      const float aq = acsh[q];
      const float4* pa = (const float4*)(CB + (long)q * CH + s0 + col);
      const float4 a0 = pa[0], a1 = pa[1];
      const float vals[8] = {a0.x, a0.y, a0.z, a0.w, a1.x, a1.y, a1.z, a1.w};
      __bf16* dst = As + row * TSTR + col;
#pragma unroll
      for (int i = 0; i < 8; ++i) {
        const int s = s0 + col + i;
        const float lm = (q >= s) ? __expf(aq - acsh[s]) : 0.f;
        dst[i] = (__bf16)(vals[i] * lm);
      }
    }
    {
      const int s = s0 + (t & 31);
      const int pq = t >> 5;
      const float dts = dth[s];
      const float4* px = (const float4*)(xh + (long)s * CONVD + p0 + pq * 16);
      const float4 x0 = px[0], x1 = px[1], x2 = px[2], x3 = px[3];
      const float vals[16] = {x0.x, x0.y, x0.z, x0.w, x1.x, x1.y, x1.z, x1.w,
                              x2.x, x2.y, x2.z, x2.w, x3.x, x3.y, x3.z, x3.w};
      const int scol = s - s0;
#pragma unroll
      for (int i = 0; i < 16; ++i)
        Bs[(pq * 16 + i) * TSTR + scol] = (__bf16)(vals[i] * dts);
    }
    __syncthreads();
    v16bf af = ld_frag(As + (wave * 16 + r) * TSTR, hh);
#pragma unroll
    for (int j = 0; j < 4; ++j) {
      v16bf bf_ = ld_frag(Bs + (j * 16 + r) * TSTR, hh);
      acc[j] = wmma_bf16(af, bf_, acc[j]);
    }
    __syncthreads();
  }
#pragma unroll
  for (int j = 0; j < 4; ++j) {
    const int p = p0 + j * 16 + r;
#pragma unroll
    for (int e = 0; e < 8; ++e) {
      const int q = q0 + wave * 16 + e + 8 * hh;
      ybuf[((long)(bc * CH + q)) * DI + h * HD + p] = acc[j][e];
    }
  }
}

// =====================================================================
// states[p,n] = sum_q (x[q,p]*dt[q]*exp(AcsLast-Acs[q])) * B[q,n], K=q=256
// =====================================================================
__global__ void states_kernel(const float* __restrict__ xbc,
                              const float* __restrict__ dtbuf,
                              const float* __restrict__ acs,
                              float* __restrict__ states) {
  __shared__ __align__(16) __bf16 As[64 * TSTR];
  __shared__ __align__(16) __bf16 Bs[64 * TSTR];
  const int z = blockIdx.z;
  const int h = z % NH, bc = z / NH;
  const int p0 = blockIdx.y * 64, n0 = blockIdx.x * 64;
  const float* acsh = acs   + ((long)bc * NH + h) * CH;
  const float* dth  = dtbuf + ((long)bc * NH + h) * CH;
  const float* xh   = xbc + (long)bc * CH * CONVD + h * HD;
  const float* Bm   = xbc + (long)bc * CH * CONVD + DI;
  const float acsLast = acsh[CH - 1];
  const int t = threadIdx.x;
  const int wave = t >> 5, lane = t & 31, r = lane & 15, hh = lane >> 4;

  v8f acc[4];
#pragma unroll
  for (int j = 0; j < 4; ++j)
#pragma unroll
    for (int e = 0; e < 8; ++e) acc[j][e] = 0.f;

  for (int q0_ = 0; q0_ < CH; q0_ += 32) {
    const int q = q0_ + (t & 31);
    const int slab = t >> 5;
    const int qcol = q - q0_;
    {
      const float wq = dth[q] * __expf(acsLast - acsh[q]);
      const float4* px = (const float4*)(xh + (long)q * CONVD + p0 + slab * 16);
      const float4 x0 = px[0], x1 = px[1], x2 = px[2], x3 = px[3];
      const float vals[16] = {x0.x, x0.y, x0.z, x0.w, x1.x, x1.y, x1.z, x1.w,
                              x2.x, x2.y, x2.z, x2.w, x3.x, x3.y, x3.z, x3.w};
#pragma unroll
      for (int i = 0; i < 16; ++i)
        As[(slab * 16 + i) * TSTR + qcol] = (__bf16)(vals[i] * wq);
    }
    {
      const float4* pb = (const float4*)(Bm + (long)q * CONVD + n0 + slab * 16);
      const float4 b0 = pb[0], b1 = pb[1], b2 = pb[2], b3 = pb[3];
      const float vals[16] = {b0.x, b0.y, b0.z, b0.w, b1.x, b1.y, b1.z, b1.w,
                              b2.x, b2.y, b2.z, b2.w, b3.x, b3.y, b3.z, b3.w};
#pragma unroll
      for (int i = 0; i < 16; ++i)
        Bs[(slab * 16 + i) * TSTR + qcol] = (__bf16)vals[i];
    }
    __syncthreads();
    v16bf af = ld_frag(As + (wave * 16 + r) * TSTR, hh);
#pragma unroll
    for (int j = 0; j < 4; ++j) {
      v16bf bf_ = ld_frag(Bs + (j * 16 + r) * TSTR, hh);
      acc[j] = wmma_bf16(af, bf_, acc[j]);
    }
    __syncthreads();
  }
  float* out = states + (long)z * HD * DSTATE;
#pragma unroll
  for (int j = 0; j < 4; ++j) {
    const int n = n0 + j * 16 + r;
#pragma unroll
    for (int e = 0; e < 8; ++e) {
      const int p = p0 + wave * 16 + e + 8 * hh;
      out[(long)p * DSTATE + n] = acc[j][e];
    }
  }
}

// =====================================================================
// Inter-chunk recurrence
// =====================================================================
__global__ void scan_kernel(const float* __restrict__ states,
                            const float* __restrict__ cdec,
                            float* __restrict__ prev) {
  const int bh = blockIdx.x;
  const int h = bh % NH, b = bh / NH;
  const int t = threadIdx.x;
  for (int i = 0; i < 64; ++i) {
    const int e = t + i * 256;
    float carry = 0.f;
    for (int c = 0; c < NCH; ++c) {
      const long off = (((long)(b * NCH + c) * NH) + h) * (HD * DSTATE) + e;
      prev[off] = carry;
      carry = carry * cdec[(b * NCH + c) * NH + h] + states[off];
    }
  }
}

// =====================================================================
// Y_off[q,p] = sum_n (C[q,n]*exp(Acs[q])) * prev[p,n];  y += Y_off + D*x
// =====================================================================
__global__ void yoff_kernel(const float* __restrict__ xbc,
                            const float* __restrict__ acs,
                            const float* __restrict__ prev,
                            const float* __restrict__ Dp,
                            float* __restrict__ ybuf) {
  __shared__ __align__(16) __bf16 As[64 * TSTR];
  __shared__ __align__(16) __bf16 Bs[64 * TSTR];
  const int z = blockIdx.z;
  const int h = z % NH, bc = z / NH;
  const int q0 = blockIdx.y * 64, p0 = blockIdx.x * 64;
  const float* acsh = acs + ((long)bc * NH + h) * CH;
  const float* Cm   = xbc + (long)bc * CH * CONVD + DI + DSTATE;
  const float* xh   = xbc + (long)bc * CH * CONVD + h * HD;
  const float* pv   = prev + (long)z * HD * DSTATE;
  const float Dh = Dp[h];
  const int t = threadIdx.x;
  const int wave = t >> 5, lane = t & 31, r = lane & 15, hh = lane >> 4;

  v8f acc[4];
#pragma unroll
  for (int j = 0; j < 4; ++j)
#pragma unroll
    for (int e = 0; e < 8; ++e) acc[j][e] = 0.f;

  for (int k0 = 0; k0 < DSTATE; k0 += 32) {
#pragma unroll
    for (int ph = 0; ph < 2; ++ph) {
      const int row = (t >> 2) + ph * 32;
      const int col = (t & 3) * 8;
      {
        const int q = q0 + row;
        const float eq = __expf(acsh[q]);
        const float4* pa = (const float4*)(Cm + (long)q * CONVD + k0 + col);
        const float4 a0 = pa[0], a1 = pa[1];
        const float vals[8] = {a0.x, a0.y, a0.z, a0.w, a1.x, a1.y, a1.z, a1.w};
        __bf16* dst = As + row * TSTR + col;
#pragma unroll
        for (int i = 0; i < 8; ++i) dst[i] = (__bf16)(vals[i] * eq);
      }
      {
        const int p = p0 + row;
        const float4* pb = (const float4*)(pv + (long)p * DSTATE + k0 + col);
        const float4 b0 = pb[0], b1 = pb[1];
        __bf16* dst = Bs + row * TSTR + col;
        dst[0] = (__bf16)b0.x; dst[1] = (__bf16)b0.y; dst[2] = (__bf16)b0.z; dst[3] = (__bf16)b0.w;
        dst[4] = (__bf16)b1.x; dst[5] = (__bf16)b1.y; dst[6] = (__bf16)b1.z; dst[7] = (__bf16)b1.w;
      }
    }
    __syncthreads();
    v16bf af = ld_frag(As + (wave * 16 + r) * TSTR, hh);
#pragma unroll
    for (int j = 0; j < 4; ++j) {
      v16bf bf_ = ld_frag(Bs + (j * 16 + r) * TSTR, hh);
      acc[j] = wmma_bf16(af, bf_, acc[j]);
    }
    __syncthreads();
  }
#pragma unroll
  for (int j = 0; j < 4; ++j) {
    const int p = p0 + j * 16 + r;
#pragma unroll
    for (int e = 0; e < 8; ++e) {
      const int q = q0 + wave * 16 + e + 8 * hh;
      const long yi = ((long)(bc * CH + q)) * DI + h * HD + p;
      const float xv = xh[(long)q * CONVD + p];
      ybuf[yi] += acc[j][e] + Dh * xv;
    }
  }
}

// =====================================================================
// y = y*silu(z);  y *= rsqrt(mean(y^2)+eps)*norm_weight  -> bf16 out
// =====================================================================
__global__ void gate_norm_kernel(const float* __restrict__ zx,
                                 const float* __restrict__ nw,
                                 const float* __restrict__ ybuf,
                                 __bf16* __restrict__ ybf) {
  const int row = blockIdx.x;              // b*LSEQ + l
  const float* zrow = zx + (long)row * NPROJ;
  const float* yrow = ybuf + (long)row * DI;
  __bf16* orow = ybf + (long)row * DI;
  __shared__ float red[256];
  float vals[8];
  float ss = 0.f;
#pragma unroll
  for (int i = 0; i < 8; ++i) {
    const int c = threadIdx.x + i * 256;
    const float yv = yrow[c] * silu_f(zrow[c]);
    vals[i] = yv;
    ss += yv * yv;
  }
  red[threadIdx.x] = ss;
  __syncthreads();
  for (int o = 128; o > 0; o >>= 1) {
    if (threadIdx.x < o) red[threadIdx.x] += red[threadIdx.x + o];
    __syncthreads();
  }
  const float scale = rsqrtf(red[0] / (float)DI + 1e-5f);
#pragma unroll
  for (int i = 0; i < 8; ++i) {
    const int c = threadIdx.x + i * 256;
    orow[c] = (__bf16)(vals[i] * scale * nw[c]);
  }
}

// =====================================================================
extern "C" void kernel_launch(void* const* d_in, const int* in_sizes, int n_in,
                              void* d_out, int out_size, void* d_ws, size_t ws_size,
                              hipStream_t stream) {
  const float* u       = (const float*)d_in[0];
  const float* W_in    = (const float*)d_in[1];
  const float* conv_w  = (const float*)d_in[2];
  const float* conv_b  = (const float*)d_in[3];
  const float* dt_bias = (const float*)d_in[4];
  const float* A_log   = (const float*)d_in[5];
  const float* Dp      = (const float*)d_in[6];
  const float* norm_w  = (const float*)d_in[7];
  const float* W_out   = (const float*)d_in[8];
  float* out = (float*)d_out;

  char* p = (char*)d_ws;
  auto alloc_f = [&](long n) { float* r = (float*)p; p += n * sizeof(float); return r; };
  auto alloc_h = [&](long n) { __bf16* r = (__bf16*)p; p += ((n + 7) & ~7L) * sizeof(__bf16); return r; };

  float* zx     = alloc_f((long)NB * LSEQ * NPROJ);
  float* xbc    = alloc_f((long)NB * LSEQ * CONVD);
  float* dtbuf  = alloc_f((long)NB * NCH * NH * CH);
  float* acsb   = alloc_f((long)NB * NCH * NH * CH);
  float* cdec   = alloc_f((long)NB * NCH * NH);
  float* cbuf   = alloc_f((long)NB * NCH * CH * CH);
  float* states = alloc_f((long)NB * NCH * NH * HD * DSTATE);
  float* prevb  = alloc_f((long)NB * NCH * NH * HD * DSTATE);
  float* ybuf   = alloc_f((long)NB * LSEQ * DI);
  // bf16 buffers (padded past M/N for unguarded tile loads)
  __bf16* u_bf    = alloc_h((long)NB * LSEQ * DM);
  __bf16* win_bf  = alloc_h((long)(NPROJ + 128) * DM);
  __bf16* wout_bf = alloc_h((long)(DM + 128) * DI);
  __bf16* xbc_bf  = alloc_h((long)NB * LSEQ * CONVD + (long)128 * CONVD);
  __bf16* ybf     = alloc_h((long)NB * LSEQ * DI);

  const int MROWS = NB * LSEQ; // 8192

  // 0) bf16 copies of GEMM operands
  f32_to_bf16_kernel<<<(unsigned)(((long)MROWS * DM) / 1024), 256, 0, stream>>>(
      u, u_bf, (long)MROWS * DM);
  f32_to_bf16_kernel<<<(unsigned)(((long)NPROJ * DM) / 1024), 256, 0, stream>>>(
      W_in, win_bf, (long)NPROJ * DM);
  f32_to_bf16_kernel<<<(unsigned)(((long)DM * DI) / 1024), 256, 0, stream>>>(
      W_out, wout_bf, (long)DM * DI);

  // 1) in-projection: zx = u @ W_in^T   (8192 x 4368, K=1024)
  gemm_nt_bf16<<<dim3((NPROJ + 127) / 128, MROWS / 128, 1), dim3(256), 0, stream>>>(
      u_bf, DM, 0L, win_bf, DM, 0L, zx, NPROJ, 0L, MROWS, NPROJ, DM);

  // 2) depthwise conv + SiLU (fp32 + bf16 outputs)
  {
    const long total = (long)NB * LSEQ * CONVD;
    conv_silu_kernel<<<(unsigned)((total + 255) / 256), 256, 0, stream>>>(
        zx, conv_w, conv_b, xbc, xbc_bf);
  }

  // 3) dt softplus + chunk cumsums
  dt_cumsum_kernel<<<2, 256, 0, stream>>>(zx, dt_bias, A_log, dtbuf, acsb, cdec);

  // 4) CB = C @ B^T per (b, chunk)   (256 x 256, K=128, 32 batches)
  gemm_nt_bf16<<<dim3(2, 2, NB * NCH), dim3(256), 0, stream>>>(
      xbc_bf + DI + DSTATE, CONVD, (long)CH * CONVD,
      xbc_bf + DI,          CONVD, (long)CH * CONVD,
      cbuf, CH, (long)CH * CH, CH, CH, DSTATE);

  // 5) Y_diag
  ydiag_kernel<<<dim3(HD / 64, CH / 64, NB * NCH * NH), dim3(128), 0, stream>>>(
      cbuf, xbc, dtbuf, acsb, ybuf);

  // 6) per-chunk states
  states_kernel<<<dim3(DSTATE / 64, HD / 64, NB * NCH * NH), dim3(128), 0, stream>>>(
      xbc, dtbuf, acsb, states);

  // 7) inter-chunk scan
  scan_kernel<<<NB * NH, 256, 0, stream>>>(states, cdec, prevb);

  // 8) Y_off + D*x
  yoff_kernel<<<dim3(HD / 64, CH / 64, NB * NCH * NH), dim3(128), 0, stream>>>(
      xbc, acsb, prevb, Dp, ybuf);

  // 9) gate + RMSNorm -> bf16
  gate_norm_kernel<<<NB * LSEQ, 256, 0, stream>>>(zx, norm_w, ybuf, ybf);

  // 10) out-projection: out = y @ W_out^T  (8192 x 1024, K=2048)
  gemm_nt_bf16<<<dim3(DM / 128, MROWS / 128, 1), dim3(256), 0, stream>>>(
      ybf, DI, 0L, wout_bf, DI, 0L, out, DM, 0L, MROWS, DM, DI);
}
